// MoCoFT_40922448396742
// MI455X (gfx1250) — matrix operand order repaired and checked by reading, not compiled, not run on hardware
//
#include <hip/hip_runtime.h>
#include <hip/hip_bf16.h>

// ---------------------------------------------------------------------------
// MoCo-style contrastive forward for MI455X (gfx1250, wave32, WMMA).
// B=1024, D=512, K=65536, NPOS=3, T=0.07
// Outputs (flat, in order): logits[1024][65537], score[1024][5],
//                           new_memory[65536][512]   (all fp32)
// Workspace: ~65.1 MB  (A_bf16 1MB @0, queue_bf16 64MB @1MB, stats @65MB)
// ---------------------------------------------------------------------------

typedef __attribute__((ext_vector_type(8)))  float  v8f;
typedef __attribute__((ext_vector_type(8)))  __bf16 v8bf;
typedef __attribute__((ext_vector_type(16))) __bf16 v16bf;

#define DIM   512
#define BQ    1024
#define KMEM  65536
#define LCOLS 65537           // 1 + K
#define INV_T (1.0f / 0.07f)

// monotone float<->uint encoding for atomicMax on signed floats
__device__ __forceinline__ unsigned fenc(float f) {
    unsigned u = __float_as_uint(f);
    return (u & 0x80000000u) ? ~u : (u | 0x80000000u);
}
__device__ __forceinline__ float fdec(unsigned u) {
    return __uint_as_float((u & 0x80000000u) ? (u ^ 0x80000000u) : ~u);
}

// ---------------------------------------------------------------------------
// Kernel 1: per-query prep. One 256-thread block (8 waves) per query row.
// Computes q=l2n(q_feat), 3 positives l2n(k_feat[perm]), hardest (argmin cos),
// mixing, score_pos. Emits A_bf16 (q_mix), new_memory[0..B), logits col0,
// and zero-inits the stats accumulators (deterministic across graph replays).
// ---------------------------------------------------------------------------
__global__ __launch_bounds__(256) void k_prep(
    const float* __restrict__ qf, const float* __restrict__ kf,
    const float* __restrict__ mask, const int* __restrict__ perm,
    __bf16* __restrict__ Abf, float* __restrict__ newmem,
    float* __restrict__ logits, float* __restrict__ spos,
    float* __restrict__ rsum, float* __restrict__ rssq,
    float* __restrict__ rcnt, unsigned* __restrict__ rmax)
{
    const int b = blockIdx.x, t = threadIdx.x;
    const int d0 = t, d1 = t + 256;
    const int lane = t & 31, wv = t >> 5;
    __shared__ float lds[8 * 8];

    const float q0 = qf[b * DIM + d0], q1 = qf[b * DIM + d1];
    float ka[3], kb[3];
#pragma unroll
    for (int j = 0; j < 3; ++j) {
        const int idx = perm[b * 3 + j];
        ka[j] = kf[(size_t)idx * DIM + d0];
        kb[j] = kf[(size_t)idx * DIM + d1];
    }
    // 7 simultaneous block reductions: |q|^2, |k_j|^2, q.k_j
    float part[8];
    part[0] = q0 * q0 + q1 * q1;
#pragma unroll
    for (int j = 0; j < 3; ++j) part[1 + j] = ka[j] * ka[j] + kb[j] * kb[j];
#pragma unroll
    for (int j = 0; j < 3; ++j) part[4 + j] = q0 * ka[j] + q1 * kb[j];
    part[7] = 0.0f;
#pragma unroll
    for (int m = 16; m >= 1; m >>= 1)
#pragma unroll
        for (int i = 0; i < 8; ++i) part[i] += __shfl_xor(part[i], m, 32);
    if (lane == 0)
#pragma unroll
        for (int i = 0; i < 8; ++i) lds[wv * 8 + i] = part[i];
    __syncthreads();
    float tot[8];
#pragma unroll
    for (int i = 0; i < 8; ++i) {
        float s = 0.0f;
#pragma unroll
        for (int w = 0; w < 8; ++w) s += lds[w * 8 + i];
        tot[i] = s;
    }

    const float nq = fmaxf(sqrtf(tot[0]), 1e-12f);
    float nk[3], sim[3];
#pragma unroll
    for (int j = 0; j < 3; ++j) {
        nk[j]  = fmaxf(sqrtf(tot[1 + j]), 1e-12f);
        sim[j] = tot[4 + j] / (nq * nk[j]);
    }
    int hj = 0;                         // argmin picks first on ties
    if (sim[1] < sim[hj]) hj = 1;
    if (sim[2] < sim[hj]) hj = 2;

    const float nkh = nk[hj];
    const float qn0 = q0 / nq,      qn1 = q1 / nq;
    const float kh0 = ka[hj] / nkh, kh1 = kb[hj] / nkh;
    const float m0 = mask[b * DIM + d0], m1 = mask[b * DIM + d1];
    const float qm0 = m0 * qn0 + (1.0f - m0) * kh0;
    const float qm1 = m1 * qn1 + (1.0f - m1) * kh1;
    const float km0 = m0 * kh0 + (1.0f - m0) * qn0;
    const float km1 = m1 * kh1 + (1.0f - m1) * qn1;

    // score_pos = sum(q_mix * k_mix)
    float sp = qm0 * km0 + qm1 * km1;
#pragma unroll
    for (int m = 16; m >= 1; m >>= 1) sp += __shfl_xor(sp, m, 32);
    __syncthreads();                    // lds reuse
    if (lane == 0) lds[wv] = sp;
    __syncthreads();
    float spb = 0.0f;
#pragma unroll
    for (int w = 0; w < 8; ++w) spb += lds[w];

    Abf[(size_t)b * DIM + d0] = (__bf16)qm0;
    Abf[(size_t)b * DIM + d1] = (__bf16)qm1;
    newmem[(size_t)b * DIM + d0] = km0;     // ring-buffer rows 0..B-1 <- k_mix
    newmem[(size_t)b * DIM + d1] = km1;
    if (t == 0) {
        logits[(size_t)b * LCOLS] = spb * INV_T;
        spos[b] = spb;
        rsum[b] = 0.0f; rssq[b] = 0.0f; rcnt[b] = 0.0f; rmax[b] = 0u;
    }
}

// ---------------------------------------------------------------------------
// Kernel 2: queue = l2norm(memory). One wave per row (8 rows / 256-thr block).
// Emits bf16 queue (GEMM B operand) and fp32 new_memory rows [B..K).
// ---------------------------------------------------------------------------
__global__ __launch_bounds__(256) void k_queue(
    const float* __restrict__ mem, __bf16* __restrict__ Qbf,
    float* __restrict__ newmem)
{
    const int row  = blockIdx.x * 8 + (threadIdx.x >> 5);
    const int lane = threadIdx.x & 31;
    const float4* p4 = (const float4*)(mem + (size_t)row * DIM + lane * 16);
    float4 a = p4[0], b = p4[1], c = p4[2], d = p4[3];
    float ssq = a.x*a.x + a.y*a.y + a.z*a.z + a.w*a.w
              + b.x*b.x + b.y*b.y + b.z*b.z + b.w*b.w
              + c.x*c.x + c.y*c.y + c.z*c.z + c.w*c.w
              + d.x*d.x + d.y*d.y + d.z*d.z + d.w*d.w;
#pragma unroll
    for (int m = 16; m >= 1; m >>= 1) ssq += __shfl_xor(ssq, m, 32);
    const float inv = 1.0f / fmaxf(sqrtf(ssq), 1e-12f);

    float v[16] = { a.x,a.y,a.z,a.w, b.x,b.y,b.z,b.w,
                    c.x,c.y,c.z,c.w, d.x,d.y,d.z,d.w };
    v8bf o0, o1;
#pragma unroll
    for (int i = 0; i < 8; ++i) { o0[i] = (__bf16)(v[i] * inv);
                                  o1[i] = (__bf16)(v[i + 8] * inv); }
    __bf16* qrow = Qbf + (size_t)row * DIM + lane * 16;
    *(v8bf*)qrow       = o0;
    *(v8bf*)(qrow + 8) = o1;

    if (row >= BQ) {                    // rows < B come from k_prep (k_mix)
        float* orow = newmem + (size_t)row * DIM + lane * 16;
        float4* o4 = (float4*)orow;
        float4 t0, t1, t2, t3;
        t0.x=v[0]*inv;  t0.y=v[1]*inv;  t0.z=v[2]*inv;  t0.w=v[3]*inv;
        t1.x=v[4]*inv;  t1.y=v[5]*inv;  t1.z=v[6]*inv;  t1.w=v[7]*inv;
        t2.x=v[8]*inv;  t2.y=v[9]*inv;  t2.z=v[10]*inv; t2.w=v[11]*inv;
        t3.x=v[12]*inv; t3.y=v[13]*inv; t3.z=v[14]*inv; t3.w=v[15]*inv;
        o4[0]=t0; o4[1]=t1; o4[2]=t2; o4[3]=t3;
    }
}

// ---------------------------------------------------------------------------
// Kernel 3: score_neg = A(bf16)[B,D] x queue(bf16)[K,D]^T via
// v_wmma_f32_16x16x32_bf16.
// Block = 8 waves (2M x 4N), block tile 64M x 256N.
// Wave tile 32M x 64N = 2 A-fragments x 4 N-subtiles = 8 accumulators, so
// every B-fragment load feeds TWO WMMAs (0.67 wmma/load) and the bf16 queue
// (64 MB, L2-resident) is swept only 16x (~1 GB L2 traffic, half of before).
// Fragment layouts used here (A and B share one contiguous k order):
//   A 16x32 bf16 : lane L holds row L%16; half (L>>4) selects K in [0,16) vs [16,32).
//   B 32x16 bf16 : lane L holds col L%16; half selects K in [0,16) vs [16,32).
//   C 16x16 f32  : vgpr r -> M = r + 8*half, N = lane%16.
// ---------------------------------------------------------------------------
__global__ __launch_bounds__(256) void k_gemm(
    const __bf16* __restrict__ A, const __bf16* __restrict__ Q,
    float* __restrict__ logits, const float* __restrict__ spos,
    float* __restrict__ rsum, float* __restrict__ rssq,
    float* __restrict__ rcnt, unsigned* __restrict__ rmax)
{
    const int t    = threadIdx.x;
    const int lane = t & 31, wv = t >> 5;
    const int wm = wv >> 2, wn = wv & 3;            // 2 x 4 wave grid
    const int m0 = blockIdx.y * 64 + wm * 32;       // wave: rows m0..m0+31
    const int n0 = blockIdx.x * 256 + wn * 64;      // wave: cols n0..n0+63
    const int col = lane & 15, half = lane >> 4;

    v8f acc[2][4];
#pragma unroll
    for (int h = 0; h < 2; ++h)
#pragma unroll
        for (int i = 0; i < 4; ++i) acc[h][i] = (v8f)0.0f;

    const __bf16* ap0 = A + (size_t)(m0 + col) * DIM;        // M rows 0..15
    const __bf16* ap1 = A + (size_t)(m0 + 16 + col) * DIM;   // M rows 16..31
    const __bf16* bp[4];
#pragma unroll
    for (int i = 0; i < 4; ++i)
        bp[i] = Q + (size_t)(n0 + i * 16 + col) * DIM + 16 * half;

#pragma unroll 2
    for (int kk = 0; kk < DIM; kk += 32) {
        v16bf afr0, afr1;
        {
            afr0 = *(const v16bf*)(ap0 + kk + 16 * half);
            afr1 = *(const v16bf*)(ap1 + kk + 16 * half);
        }
        if (kk + 64 < DIM) {            // stream B ahead (global_prefetch_b8)
            __builtin_prefetch(bp[0] + kk + 64, 0, 2);
            __builtin_prefetch(bp[2] + kk + 64, 0, 2);
        }
#pragma unroll
        for (int i = 0; i < 4; ++i) {
            const v16bf bfr = *(const v16bf*)(bp[i] + kk);
            acc[0][i] = __builtin_amdgcn_wmma_f32_16x16x32_bf16(
                false, afr0, false, bfr, (short)0, acc[0][i], false, false);
            acc[1][i] = __builtin_amdgcn_wmma_f32_16x16x32_bf16(
                false, afr1, false, bfr, (short)0, acc[1][i], false, false);
        }
    }

    // ---- logits = score_neg / T (column 0 is score_pos, so +1 shift) ----
#pragma unroll
    for (int h = 0; h < 2; ++h)
#pragma unroll
    for (int i = 0; i < 4; ++i) {
        const size_t colg = (size_t)(1 + n0 + i * 16 + col);
#pragma unroll
        for (int r = 0; r < 8; ++r) {
            const int rowg = m0 + h * 16 + half * 8 + r;
            logits[(size_t)rowg * LCOLS + colg] = acc[h][i][r] * INV_T;
        }
    }

    // ---- per-row stats on raw score_neg: sum, sumsq, count>sp, max ----
#pragma unroll
    for (int h = 0; h < 2; ++h)
#pragma unroll
    for (int r = 0; r < 8; ++r) {
        const int rowg = m0 + h * 16 + half * 8 + r;
        const float sp = spos[rowg];
        float s = 0.0f, ss = 0.0f, c = 0.0f, mx = -__builtin_inff();
#pragma unroll
        for (int i = 0; i < 4; ++i) {
            const float v = acc[h][i][r];
            s += v; ss += v * v;
            c += (v > sp) ? 1.0f : 0.0f;
            mx = fmaxf(mx, v);
        }
        // reduce across the 16 lanes sharing this row (xor<=8 stays in half)
#pragma unroll
        for (int m = 8; m >= 1; m >>= 1) {
            s  += __shfl_xor(s, m, 32);
            ss += __shfl_xor(ss, m, 32);
            c  += __shfl_xor(c, m, 32);
            mx  = fmaxf(mx, __shfl_xor(mx, m, 32));
        }
        if (col == 0) {
            atomicAdd(&rsum[rowg], s);
            atomicAdd(&rssq[rowg], ss);
            atomicAdd(&rcnt[rowg], c);
            atomicMax(&rmax[rowg], fenc(mx));
        }
    }
}

// ---------------------------------------------------------------------------
// Kernel 4: finalize score[B][5] = mean, var(ddof=1), count_neg, count_pos, sp
// ---------------------------------------------------------------------------
__global__ __launch_bounds__(256) void k_stats(
    const float* __restrict__ rsum, const float* __restrict__ rssq,
    const float* __restrict__ rcnt, const unsigned* __restrict__ rmax,
    const float* __restrict__ spos, float* __restrict__ score)
{
    const int b = blockIdx.x * 256 + threadIdx.x;
    if (b >= BQ) return;
    const float s = rsum[b], ss = rssq[b], c = rcnt[b], sp = spos[b];
    const float mean = s * (1.0f / (float)KMEM);
    const float var  = (ss - s * mean) * (1.0f / (float)(KMEM - 1));
    const float mx   = fdec(rmax[b]);
    score[b * 5 + 0] = mean;
    score[b * 5 + 1] = var;
    score[b * 5 + 2] = c;
    score[b * 5 + 3] = (mx > sp) ? 1.0f : 0.0f;
    score[b * 5 + 4] = sp;
}

// ---------------------------------------------------------------------------
extern "C" void kernel_launch(void* const* d_in, const int* in_sizes, int n_in,
                              void* d_out, int out_size, void* d_ws, size_t ws_size,
                              hipStream_t stream)
{
    const float* qf   = (const float*)d_in[0];   // [1024,512]
    const float* kf   = (const float*)d_in[1];   // [3072,512]
    const float* mem  = (const float*)d_in[2];   // [65536,512]
    const float* mask = (const float*)d_in[3];   // [1024,512]
    const int*   perm = (const int*)  d_in[4];   // [3072]

    float* out    = (float*)d_out;
    float* logits = out;                                   // [1024][65537]
    float* score  = out + (size_t)BQ * LCOLS;              // [1024][5]
    float* newmem = score + (size_t)BQ * 5;                // [65536][512]

    char* ws = (char*)d_ws;                                // needs ~65.1 MB
    __bf16*   Abf  = (__bf16*)ws;                          // 1 MB
    __bf16*   Qbf  = (__bf16*)(ws + (1u << 20));           // 64 MB
    char*     sb   = ws + (1u << 20) + (64u << 20);
    float*    spos = (float*)sb;                           // 4 KB
    float*    rsum = spos + BQ;
    float*    rssq = rsum + BQ;
    float*    rcnt = rssq + BQ;
    unsigned* rmax = (unsigned*)(rcnt + BQ);

    k_prep <<<BQ, 256, 0, stream>>>(qf, kf, mask, perm, Abf, newmem,
                                    logits, spos, rsum, rssq, rcnt, rmax);
    k_queue<<<KMEM / 8, 256, 0, stream>>>(mem, Qbf, newmem);
    k_gemm <<<dim3(KMEM / 256, BQ / 64), 256, 0, stream>>>(
        Abf, Qbf, logits, spos, rsum, rssq, rcnt, rmax);
    k_stats<<<BQ / 256, 256, 0, stream>>>(rsum, rssq, rcnt, rmax, spos, score);
}
